// _ScratchXLSTM_6640019439853
// MI455X (gfx1250) — matrix operand order repaired and checked by reading, not compiled, not run on hardware
//
#include <hip/hip_runtime.h>
#include <hip/hip_bf16.h>
#include <math.h>

// Model dims
#define BTN 65536   // B*T
#define DM  128
#define HN  4
#define HD  32
#define TN  64
#define VN  9

typedef __attribute__((ext_vector_type(16))) __bf16       v16bf;
typedef __attribute__((ext_vector_type(8)))  float        v8f;
typedef __attribute__((ext_vector_type(4)))  unsigned int u32x4;

union FragB { v16bf v; u32x4 u[2]; };

__device__ __forceinline__ float wave_sum32(float v) {
#pragma unroll
  for (int off = 16; off; off >>= 1) v += __shfl_xor(v, off, 32);
  return v;
}

// ---------------- embedding gather ----------------
__global__ void k_embed(const int* __restrict__ x, const float* __restrict__ emb,
                        float* __restrict__ h) {
  int i = blockIdx.x * blockDim.x + threadIdx.x;
  int row = i >> 7, d = i & 127;
  h[i] = emb[x[row] * DM + d];
}

// ---------------- LayerNorm (f32 in -> bf16 out) ----------------
__global__ void k_layernorm_bf16(const float* __restrict__ in, const float* __restrict__ s,
                                 const float* __restrict__ b, __bf16* __restrict__ out) {
  int row = blockIdx.x, tid = threadIdx.x;            // blockDim = 128
  float v = in[(size_t)row * DM + tid];
  float sum = wave_sum32(v);
  float sq  = wave_sum32(v * v);
  __shared__ float ws[4], wq[4];
  int wid = tid >> 5, lane = tid & 31;
  if (!lane) { ws[wid] = sum; wq[wid] = sq; }
  __syncthreads();
  sum = ws[0] + ws[1] + ws[2] + ws[3];
  sq  = wq[0] + wq[1] + wq[2] + wq[3];
  float mu  = sum * (1.0f / DM);
  float var = sq * (1.0f / DM) - mu * mu;
  float r   = rsqrtf(var + 1e-5f);
  out[(size_t)row * DM + tid] = (__bf16)((v - mu) * r * s[tid] + b[tid]);
}

// ---------------- pack f32 weights into WMMA B-fragment order ----------------
// dst[kt][nt][lane][0..15] = W[kt*32 + (lane>>4)*16 + j][nt*16 + (lane&15)]  (bf16)
__global__ void k_pack_w(const float* __restrict__ W, __bf16* __restrict__ dst,
                         int K, int N) {
  int t = blockIdx.x * blockDim.x + threadIdx.x;
  int ntiles = N >> 4;
  int total = (K >> 5) * ntiles * 32;
  if (t >= total) return;
  int lane = t & 31, tile = t >> 5;
  int nt = tile % ntiles, kt = tile / ntiles;
  int col = lane & 15, khalf = lane >> 4;
  int kbase = kt * 32 + khalf * 16;
  __bf16* d = dst + ((size_t)tile * 32 + lane) * 16;
  const float* src = W + (size_t)kbase * N + nt * 16 + col;
#pragma unroll
  for (int j = 0; j < 16; ++j) d[j] = (__bf16)src[(size_t)j * N];
}

// ---------------- WMMA GEMM, 16x64 register-blocked per wave ----------------
// C[M,N] = A[M,K](bf16) * Wpk + bias
// mode 0: Cf = v        (gate pre-activations)
// mode 1: Cb = gelu(v)  (MLP hidden, bf16 for next GEMM)
// mode 2: Cf += v       (MLP output + residual)
__global__ void k_gemm_bf16_wmma(const __bf16* __restrict__ A, const __bf16* __restrict__ Wpk,
                                 const float* __restrict__ bias, float* __restrict__ Cf,
                                 __bf16* __restrict__ Cb, int M, int N, int K, int mode) {
  int wid  = blockIdx.x * (blockDim.x >> 5) + (threadIdx.x >> 5);
  int lane = threadIdx.x & 31;
  int ntiles = N >> 4;
  int ngrp   = N >> 6;                 // groups of 4 N-tiles (64 cols)
  int mt = wid / ngrp, ng = wid % ngrp;
  if (mt * 16 >= M) return;
  int nt0 = ng * 4;
  int khalf = lane >> 4;
  const __bf16* arow = A + (size_t)(mt * 16 + (lane & 15)) * K;
  v8f acc0 = {}, acc1 = {}, acc2 = {}, acc3 = {};
  for (int kt = 0; kt < K; kt += 32) {
    FragB a;
    // 16-bit A-matrix 16x32 layout (contiguous, matches packed B): VGPR0-7 <- K = khalf*16 .. +15
    a.u[0] = *(const u32x4*)(const void*)(arow + kt + khalf * 16);
    a.u[1] = *(const u32x4*)(const void*)(arow + kt + khalf * 16 + 8);
    if (kt + 32 < K) __builtin_prefetch(arow + kt + 32, 0, 1);   // global_prefetch
    const __bf16* wp = Wpk + ((size_t)((kt >> 5) * ntiles + nt0) * 32 + lane) * 16;
    FragB b0, b1, b2, b3;
    b0.u[0] = *(const u32x4*)(const void*)(wp + 0 * 512);
    b0.u[1] = *(const u32x4*)(const void*)(wp + 0 * 512 + 8);
    b1.u[0] = *(const u32x4*)(const void*)(wp + 1 * 512);
    b1.u[1] = *(const u32x4*)(const void*)(wp + 1 * 512 + 8);
    b2.u[0] = *(const u32x4*)(const void*)(wp + 2 * 512);
    b2.u[1] = *(const u32x4*)(const void*)(wp + 2 * 512 + 8);
    b3.u[0] = *(const u32x4*)(const void*)(wp + 3 * 512);
    b3.u[1] = *(const u32x4*)(const void*)(wp + 3 * 512 + 8);
    acc0 = __builtin_amdgcn_wmma_f32_16x16x32_bf16(false, a.v, false, b0.v, (short)0, acc0, false, false);
    acc1 = __builtin_amdgcn_wmma_f32_16x16x32_bf16(false, a.v, false, b1.v, (short)0, acc1, false, false);
    acc2 = __builtin_amdgcn_wmma_f32_16x16x32_bf16(false, a.v, false, b2.v, (short)0, acc2, false, false);
    acc3 = __builtin_amdgcn_wmma_f32_16x16x32_bf16(false, a.v, false, b3.v, (short)0, acc3, false, false);
  }
  // C/D layout: VGPR r -> (M = khalf*8 + r, N = lane&15)
  int rbase = mt * 16 + khalf * 8;
  v8f accs[4] = {acc0, acc1, acc2, acc3};
#pragma unroll
  for (int j = 0; j < 4; ++j) {
    int colg = (nt0 + j) * 16 + (lane & 15);
    float bv = bias[colg];
#pragma unroll
    for (int r = 0; r < 8; ++r) {
      float v = accs[j][r] + bv;
      size_t idx = (size_t)(rbase + r) * N + colg;
      if (mode == 0) {
        Cf[idx] = v;
      } else if (mode == 1) {
        float g = 0.5f * v * (1.0f + erff(v * 0.70710678118654752f));
        Cb[idx] = (__bf16)g;
      } else {
        Cf[idx] += v;
      }
    }
  }
}

// ---------------- sLSTM sequential scan: one wave32 per (batch, head) ----------------
__global__ void k_slstm(const float* __restrict__ gx, const float* __restrict__ R,
                        float* __restrict__ sout) {
  int b    = blockIdx.x >> 2;
  int head = blockIdx.x & 3;
  int lane = threadIdx.x;                 // e = 0..31
  __shared__ float Rl[4 * HD * HD];       // [gate][d][e], 16KB
  __shared__ float hl[HD];
  for (int i = lane; i < 4 * HD * HD; i += 32) {
    int g = i >> 10, rem = i & 1023;      // rem = d*32 + e
    Rl[i] = R[(size_t)g * (HN * HD * HD) + (size_t)head * (HD * HD) + rem];
  }
  float c = 0.f, n = 0.f, m = 0.f, hst = 0.f;
  __syncthreads();
  for (int t = 0; t < TN; ++t) {
    hl[lane] = hst;
    __syncthreads();
    float r0 = 0.f, r1 = 0.f, r2 = 0.f, r3 = 0.f;
#pragma unroll 8
    for (int d = 0; d < HD; ++d) {
      float hd = hl[d];
      r0 = fmaf(hd, Rl[0 * 1024 + d * 32 + lane], r0);
      r1 = fmaf(hd, Rl[1 * 1024 + d * 32 + lane], r1);
      r2 = fmaf(hd, Rl[2 * 1024 + d * 32 + lane], r2);
      r3 = fmaf(hd, Rl[3 * 1024 + d * 32 + lane], r3);
    }
    size_t gb = ((size_t)b * TN + t) * (4 * DM) + head * HD + lane;
    float it = gx[gb + 0 * DM] + r0;
    float ft = gx[gb + 1 * DM] + r1;
    float zt = gx[gb + 2 * DM] + r2;
    float ot = gx[gb + 3 * DM] + r3;
    float mn = fmaxf(ft + m, it);
    float ii = __expf(it - mn);
    float ff = __expf(ft + m - mn);
    c = ff * c + ii * tanhf(zt);
    n = ff * n + ii;
    m = mn;
    float sg = 1.0f / (1.0f + __expf(-ot));
    hst = sg * c / (n + 1e-6f);
    sout[((size_t)b * TN + t) * DM + head * HD + lane] = hst;
    __syncthreads();
  }
}

// ---------------- per-head GroupNorm + residual (heads align with wave32) ----------------
__global__ void k_gnres(const float* __restrict__ sv, const float* __restrict__ gs,
                        const float* __restrict__ gb, float* __restrict__ h) {
  int row = blockIdx.x, tid = threadIdx.x;   // blockDim = 128, wave = head
  float v = sv[(size_t)row * DM + tid];
  float sum = wave_sum32(v);
  float sq  = wave_sum32(v * v);
  float mu  = sum * (1.0f / HD);
  float var = sq * (1.0f / HD) - mu * mu;
  float r   = rsqrtf(var + 1e-5f);
  h[(size_t)row * DM + tid] += (v - mu) * r * gs[tid] + gb[tid];
}

// ---------------- final LN + projection D -> V=9 ----------------
__global__ void k_final_proj(const float* __restrict__ h, const float* __restrict__ pns,
                             const float* __restrict__ pnb, const float* __restrict__ pw,
                             const float* __restrict__ pb, float* __restrict__ out) {
  int row = blockIdx.x, tid = threadIdx.x;   // blockDim = 128
  __shared__ float hn[DM];
  __shared__ float ws[4], wq[4];
  float v = h[(size_t)row * DM + tid];
  float sum = wave_sum32(v);
  float sq  = wave_sum32(v * v);
  int wid = tid >> 5, lane = tid & 31;
  if (!lane) { ws[wid] = sum; wq[wid] = sq; }
  __syncthreads();
  sum = ws[0] + ws[1] + ws[2] + ws[3];
  sq  = wq[0] + wq[1] + wq[2] + wq[3];
  float mu  = sum * (1.0f / DM);
  float var = sq * (1.0f / DM) - mu * mu;
  float r   = rsqrtf(var + 1e-5f);
  hn[tid] = (v - mu) * r * pns[tid] + pnb[tid];
  __syncthreads();
  if (tid < VN) {
    float acc = pb[tid];
#pragma unroll 16
    for (int d = 0; d < DM; ++d) acc = fmaf(hn[d], pw[d * VN + tid], acc);
    out[(size_t)row * VN + tid] = acc;
  }
}

extern "C" void kernel_launch(void* const* d_in, const int* in_sizes, int n_in,
                              void* d_out, int out_size, void* d_ws, size_t ws_size,
                              hipStream_t stream) {
  const int*   x    = (const int*)d_in[0];
  const float* emb  = (const float*)d_in[1];
  const float* ln1s = (const float*)d_in[2];
  const float* ln1b = (const float*)d_in[3];
  const float* Wg   = (const float*)d_in[4];
  const float* bg   = (const float*)d_in[5];
  const float* R    = (const float*)d_in[6];
  const float* gns  = (const float*)d_in[7];
  const float* gnb  = (const float*)d_in[8];
  const float* ln2s = (const float*)d_in[9];
  const float* ln2b = (const float*)d_in[10];
  const float* W1   = (const float*)d_in[11];
  const float* b1   = (const float*)d_in[12];
  const float* W2   = (const float*)d_in[13];
  const float* b2   = (const float*)d_in[14];
  const float* pns  = (const float*)d_in[15];
  const float* pnb  = (const float*)d_in[16];
  const float* pw   = (const float*)d_in[17];
  const float* pb   = (const float*)d_in[18];
  float* out = (float*)d_out;

  // workspace carve-up
  char* ws = (char*)d_ws;
  size_t o = 0;
  float*  h    = (float*)(ws + o);  o += (size_t)BTN * DM * sizeof(float);     // 33.5 MB
  __bf16* xbf  = (__bf16*)(ws + o); o += (size_t)BTN * DM * sizeof(__bf16);    // 16.8 MB
  float*  gx   = (float*)(ws + o);  o += (size_t)BTN * 4 * DM * sizeof(float); // 134 MB
  float*  sout = (float*)(ws + o);  o += (size_t)BTN * DM * sizeof(float);     // 33.5 MB
  __bf16* a1   = (__bf16*)(ws + o); o += (size_t)BTN * DM * sizeof(__bf16);    // 16.8 MB
  __bf16* wgp  = (__bf16*)(ws + o); o += (size_t)DM * 4 * DM * sizeof(__bf16);
  __bf16* w1p  = (__bf16*)(ws + o); o += (size_t)DM * DM * sizeof(__bf16);
  __bf16* w2p  = (__bf16*)(ws + o); o += (size_t)DM * DM * sizeof(__bf16);
  (void)ws_size; (void)n_in; (void)in_sizes; (void)out_size;

  k_embed<<<(BTN * DM) / 256, 256, 0, stream>>>(x, emb, h);

  for (int blk = 0; blk < 2; ++blk) {
    // LN1 -> bf16
    k_layernorm_bf16<<<BTN, 128, 0, stream>>>(h, ln1s + blk * DM, ln1b + blk * DM, xbf);
    // gate GEMM: (BTNx128) x (128x512); 16x64 tile per wave
    k_pack_w<<<16, 256, 0, stream>>>(Wg + (size_t)blk * DM * 4 * DM, wgp, DM, 4 * DM);
    k_gemm_bf16_wmma<<<(BTN / 16) * (4 * DM / 64) / 8, 256, 0, stream>>>(
        xbf, wgp, bg + blk * 4 * DM, gx, nullptr, BTN, 4 * DM, DM, 0);
    // recurrent scan
    k_slstm<<<1024 * HN, 32, 0, stream>>>(gx, R + (size_t)blk * 4 * HN * HD * HD, sout);
    // groupnorm + residual
    k_gnres<<<BTN, 128, 0, stream>>>(sout, gns + blk * DM, gnb + blk * DM, h);
    // MLP
    k_layernorm_bf16<<<BTN, 128, 0, stream>>>(h, ln2s + blk * DM, ln2b + blk * DM, xbf);
    k_pack_w<<<4, 256, 0, stream>>>(W1 + (size_t)blk * DM * DM, w1p, DM, DM);
    k_gemm_bf16_wmma<<<(BTN / 16) * (DM / 64) / 8, 256, 0, stream>>>(
        xbf, w1p, b1 + blk * DM, nullptr, a1, BTN, DM, DM, 1);
    k_pack_w<<<4, 256, 0, stream>>>(W2 + (size_t)blk * DM * DM, w2p, DM, DM);
    k_gemm_bf16_wmma<<<(BTN / 16) * (DM / 64) / 8, 256, 0, stream>>>(
        a1, w2p, b2 + blk * DM, h, nullptr, BTN, DM, DM, 2);
  }

  k_final_proj<<<BTN, 128, 0, stream>>>(h, pns, pnb, pw, pb, out);
}
